// MyT5Attention_65111704207697
// MI455X (gfx1250) — compile-verified, matrix operand order checked by reading, not run on hardware
//
#include <hip/hip_runtime.h>
#include <hip/hip_bf16.h>

// ---------------------------------------------------------------------------
// Types for WMMA fragments (CDNA5 / gfx1250, wave32)
// ---------------------------------------------------------------------------
typedef __attribute__((ext_vector_type(16))) __bf16 v16bf;
typedef __attribute__((ext_vector_type(8)))  float  v8f;
typedef __attribute__((ext_vector_type(4)))  unsigned int v4u;

union FragU {
    v16bf v;
    v4u   q[2];
    unsigned short u[16];
};

static __device__ __forceinline__ unsigned short f2bf(float f) {
    unsigned int u = __float_as_uint(f);
    unsigned int r = u + 0x7FFFu + ((u >> 16) & 1u);   // round-to-nearest-even
    return (unsigned short)(r >> 16);
}

static __device__ __forceinline__ v8f wmma_bf16(v16bf a, v16bf b, v8f c) {
    return __builtin_amdgcn_wmma_f32_16x16x32_bf16(
        /*neg_a=*/false, a, /*neg_b=*/false, b,
        /*c_mod=*/(short)0, c, /*reuse_a=*/false, /*reuse_b=*/false);
}

// Async global->LDS 16B copies (gfx1250 GLOBAL_LOAD_ASYNC_TO_LDS_B128,
// tracked by ASYNCcnt). LDS address = low 32 bits of the generic pointer
// (ISA flat->LDS truncation rule).
static __device__ __forceinline__ void async_b128(void* lds, const void* g) {
    unsigned loff = (unsigned)(unsigned long long)(uintptr_t)lds;
    unsigned long long ga = (unsigned long long)(uintptr_t)g;
    asm volatile("global_load_async_to_lds_b128 %0, %1, off"
                 :: "v"(loff), "v"(ga) : "memory");
}
// Non-temporal variant for stream-once data (bias): keep it out of L2's
// working set so K/V stay resident.
static __device__ __forceinline__ void async_b128_nt(void* lds, const void* g) {
    unsigned loff = (unsigned)(unsigned long long)(uintptr_t)lds;
    unsigned long long ga = (unsigned long long)(uintptr_t)g;
    asm volatile("global_load_async_to_lds_b128 %0, %1, off th:TH_LOAD_NT"
                 :: "v"(loff), "v"(ga) : "memory");
}
static __device__ __forceinline__ void wait_async0() {
    asm volatile("s_wait_asynccnt 0" ::: "memory");
}

// Problem constants
#define BB 2
#define SS 2048
#define DD 1024
#define HH 16
#define DK 64
#define PP 64
#define LL 2112          // P + S
#define BS 4096          // B * S
#define NC (LL / 32)     // 66 key chunks

// ---------------------------------------------------------------------------
// Elementwise fp32 -> bf16 cast
// ---------------------------------------------------------------------------
__global__ void cast_bf16_kernel(const float* __restrict__ in,
                                 unsigned short* __restrict__ out, int n) {
    int i = blockIdx.x * blockDim.x + threadIdx.x;
    if (i < n) out[i] = f2bf(in[i]);
}

// fp32 [K,N] -> bf16 [N,K] transpose-cast (LDS tiled, 32x32 tiles)
__global__ __launch_bounds__(256) void castT_kernel(
        const float* __restrict__ in, unsigned short* __restrict__ out) {
    __shared__ float T[32][33];
    const int k0 = blockIdx.y * 32;
    const int n0 = blockIdx.x * 32;
    const int tr = threadIdx.x >> 5;    // 0..7
    const int tc = threadIdx.x & 31;
#pragma unroll
    for (int i = 0; i < 4; ++i)
        T[tr + i * 8][tc] = in[(k0 + tr + i * 8) * DD + n0 + tc];
    __syncthreads();
#pragma unroll
    for (int i = 0; i < 4; ++i)
        out[(n0 + tr + i * 8) * DD + k0 + tc] = f2bf(T[tc][tr + i * 8]);
}

// kv_prompt_k [B,H,P,dk] fp32 -> Kb [B,H,L,dk] bf16 at rows [0,P)
__global__ void prompt_k_kernel(const float* __restrict__ in,
                                unsigned short* __restrict__ Kb) {
    int i = blockIdx.x * blockDim.x + threadIdx.x;
    if (i >= BB * HH * PP * DK) return;
    int d = i & 63, p = (i >> 6) & 63, h = (i >> 12) & 15, b = i >> 16;
    Kb[(((b * HH + h) * LL) + p) * DK + d] = f2bf(in[i]);
}

// kv_prompt_v [B,H,P,dk] fp32 -> Vt [B,H,dk,L] bf16 at cols [0,P)
__global__ void prompt_v_kernel(const float* __restrict__ in,
                                unsigned short* __restrict__ Vt) {
    int i = blockIdx.x * blockDim.x + threadIdx.x;
    if (i >= BB * HH * PP * DK) return;
    int d = i & 63, p = (i >> 6) & 63, h = (i >> 12) & 15, b = i >> 16;
    Vt[(((b * HH + h) * DK) + d) * LL + p] = f2bf(in[i]);
}

// ---------------------------------------------------------------------------
// Tiled bf16 WMMA GEMM:  C[4096,1024] = A[4096,1024] * W[1024,1024]
// W pre-transposed (Wt[N,K]) so both tiles stage via contiguous 16B async
// global->LDS copies, double-buffered; pipeline peeled so the steady-state
// loop has no staging guard.
// Block: 256 threads (8 waves), tile 128x128, K-step 32.
// mode: 0 Q[B,H,S,dk] | 1 K[B,H,L,dk]+P | 2 V^T[B,H,dk,L]+P | 3 fp32 out
// ---------------------------------------------------------------------------
#define LDSA_STRIDE 40   // 32 cols + 8 pad (keeps 16B alignment; klo in {0,8})

__global__ __launch_bounds__(256) void gemm_bf16_kernel(
        const unsigned short* __restrict__ A,
        const unsigned short* __restrict__ Wt,
        void* __restrict__ outp, int mode) {
    __shared__ __align__(16) unsigned short As[2][128 * LDSA_STRIDE];
    __shared__ __align__(16) unsigned short Bt[2][128 * LDSA_STRIDE];

    const int t    = threadIdx.x;
    const int lane = t & 31;
    const int wave = t >> 5;
    const int wm   = wave >> 2;          // 0..1
    const int wn   = wave & 3;           // 0..3
    const int half = lane >> 4;          // 0..1
    const int l15  = lane & 15;
    const int klo  = half ? 8 : 0;

    const int m0 = blockIdx.y * 128;
    const int n0 = blockIdx.x * 128;

    v8f acc[4][2];
#pragma unroll
    for (int mt = 0; mt < 4; ++mt)
#pragma unroll
        for (int nt = 0; nt < 2; ++nt)
#pragma unroll
            for (int r = 0; r < 8; ++r) acc[mt][nt][r] = 0.0f;

    auto stage = [&](int buf, int k0) {
#pragma unroll
        for (int c = t; c < 512; c += 256) {
            int row = c >> 2, ch = c & 3;
            async_b128(&As[buf][row * LDSA_STRIDE + ch * 8],
                       &A[(m0 + row) * DD + k0 + ch * 8]);
            async_b128(&Bt[buf][row * LDSA_STRIDE + ch * 8],
                       &Wt[(n0 + row) * DD + k0 + ch * 8]);
        }
    };

    auto compute = [&](int buf) {
        v16bf af[4], bfr[2];
#pragma unroll
        for (int mt = 0; mt < 4; ++mt) {
            FragU f;
            int row = wm * 64 + mt * 16 + l15;
            f.q[0] = *reinterpret_cast<const v4u*>(&As[buf][row * LDSA_STRIDE + klo]);
            f.q[1] = *reinterpret_cast<const v4u*>(&As[buf][row * LDSA_STRIDE + klo + 16]);
            af[mt] = f.v;
        }
#pragma unroll
        for (int nt = 0; nt < 2; ++nt) {
            FragU f;
            int col = wn * 32 + nt * 16 + l15;
            f.q[0] = *reinterpret_cast<const v4u*>(&Bt[buf][col * LDSA_STRIDE + klo]);
            f.q[1] = *reinterpret_cast<const v4u*>(&Bt[buf][col * LDSA_STRIDE + klo + 16]);
            bfr[nt] = f.v;
        }
#pragma unroll
        for (int mt = 0; mt < 4; ++mt)
#pragma unroll
            for (int nt = 0; nt < 2; ++nt)
                acc[mt][nt] = wmma_bf16(af[mt], bfr[nt], acc[mt][nt]);
    };

    stage(0, 0);
    wait_async0();
    __syncthreads();

    int cur = 0;
    for (int k0 = 0; k0 < DD - 32; k0 += 32) {   // steady state: always stage
        stage(cur ^ 1, k0 + 32);
        compute(cur);
        wait_async0();
        __syncthreads();
        cur ^= 1;
    }
    compute(cur);                                 // last tile, no staging

    // Write out. C tile layout: VGPR r -> (M = r + half*8, N = l15)
#pragma unroll
    for (int mt = 0; mt < 4; ++mt)
#pragma unroll
        for (int nt = 0; nt < 2; ++nt)
#pragma unroll
            for (int r = 0; r < 8; ++r) {
                int row = m0 + wm * 64 + mt * 16 + r + half * 8;   // 0..4095
                int col = n0 + wn * 32 + nt * 16 + l15;            // 0..1023
                float val = acc[mt][nt][r];
                if (mode == 3) {
                    reinterpret_cast<float*>(outp)[row * DD + col] = val;
                } else {
                    unsigned short bv = f2bf(val);
                    unsigned short* o = reinterpret_cast<unsigned short*>(outp);
                    int b = row >> 11, s = row & 2047;
                    int h = col >> 6,  d = col & 63;
                    if (mode == 0)
                        o[(((b * HH + h) * SS) + s) * DK + d] = bv;
                    else if (mode == 1)
                        o[(((b * HH + h) * LL) + PP + s) * DK + d] = bv;
                    else
                        o[(((b * HH + h) * DK) + d) * LL + PP + s] = bv;
                }
            }
}

// ---------------------------------------------------------------------------
// Flash-style attention. Block = 4 waves sharing one (b,h); K/V chunks AND
// per-wave bias tiles are staged into LDS with double-buffered async copies.
// Each wave owns one 16-query tile. blockDim = 128, grid = B*H*(S/64) = 1024
// ---------------------------------------------------------------------------
#define KL_STRIDE 72    // 64 dk + 8 pad
#define VL_STRIDE 40    // 32 keys + 8 pad

__global__ __launch_bounds__(128) void attn_kernel(
        const unsigned short* __restrict__ Qb,   // [B,H,S,dk]
        const unsigned short* __restrict__ Kb,   // [B,H,L,dk]
        const unsigned short* __restrict__ Vt,   // [B,H,dk,L]
        const float* __restrict__ bias,          // [1,H,S,L]
        unsigned short* __restrict__ Ctx) {      // [B,S,H*dk]
    __shared__ __align__(16) unsigned short Kl[2][32 * KL_STRIDE];
    __shared__ __align__(16) unsigned short Vl[2][64 * VL_STRIDE];
    __shared__ __align__(16) float          Bl[2][4][16 * 32];
    __shared__ __align__(16) unsigned short Pl[4 * 512];

    const int t    = threadIdx.x;
    const int lane = t & 31;
    const int wv   = t >> 5;
    const int blk  = blockIdx.x;                  // 0..1023
    const int b    = blk >> 9;
    const int h    = (blk >> 5) & 15;
    const int qq   = blk & 31;
    const int qbase = qq * 64 + wv * 16;

    const int half = lane >> 4;
    const int l15  = lane & 15;
    const int klo  = half ? 8 : 0;

    unsigned short* P = &Pl[wv * 512];

    const unsigned short* kbh = &Kb[(size_t)(b * HH + h) * LL * DK];
    const unsigned short* vbh = &Vt[(size_t)(b * HH + h) * DK * LL];
    const float*          bh  = &bias[(size_t)(h * SS + qbase) * LL];

    // K/V chunks staged block-cooperatively; bias tile staged per-wave.
    auto stageAll = [&](int buf, int key0) {
#pragma unroll
        for (int c = t; c < 256; c += 128) {
            int kr = c >> 3, kch = c & 7;          // K: 32 rows x 8 chunks
            async_b128(&Kl[buf][kr * KL_STRIDE + kch * 8],
                       &kbh[(size_t)(key0 + kr) * DK + kch * 8]);
            int vr = c >> 2, vch = c & 3;          // V: 64 rows x 4 chunks
            async_b128(&Vl[buf][vr * VL_STRIDE + vch * 8],
                       &vbh[(size_t)vr * LL + key0 + vch * 8]);
        }
        float* bd = &Bl[buf][wv][0];
#pragma unroll
        for (int c = lane; c < 128; c += 32) {     // bias: 16 rows x 8 chunks
            int rr = c >> 3, ch = c & 7;
            async_b128_nt(&bd[rr * 32 + ch * 4],
                          &bh[(size_t)(key0 + rr * LL) + ch * 4]);
        }
    };

    // Q fragments (A-matrix 16x64, two K=32 chunks)
    v16bf qf[2];
    {
        const unsigned short* qp =
            &Qb[((size_t)(b * HH + h) * SS + qbase + l15) * DK];
#pragma unroll
        for (int c = 0; c < 2; ++c) {
            FragU f;
            f.q[0] = *reinterpret_cast<const v4u*>(&qp[c * 32 + klo]);
            f.q[1] = *reinterpret_cast<const v4u*>(&qp[c * 32 + klo + 16]);
            qf[c] = f.v;
        }
    }

    v8f acc[4];
#pragma unroll
    for (int j = 0; j < 4; ++j)
#pragma unroll
        for (int r = 0; r < 8; ++r) acc[j][r] = 0.0f;
    float m[8], l[8];
#pragma unroll
    for (int r = 0; r < 8; ++r) { m[r] = -3.0e38f; l[r] = 0.0f; }

    auto body = [&](int cur) {
        // K fragments from LDS: B-matrix K^T[64 x 16keys]
        v16bf kf[2][2];
#pragma unroll
        for (int tkt = 0; tkt < 2; ++tkt) {
            const unsigned short* kp = &Kl[cur][(tkt * 16 + l15) * KL_STRIDE];
#pragma unroll
            for (int cc = 0; cc < 2; ++cc) {
                FragU f;
                f.q[0] = *reinterpret_cast<const v4u*>(&kp[cc * 32 + klo]);
                f.q[1] = *reinterpret_cast<const v4u*>(&kp[cc * 32 + klo + 16]);
                kf[tkt][cc] = f.v;
            }
        }

        v8f z;
#pragma unroll
        for (int r = 0; r < 8; ++r) z[r] = 0.0f;
        v8f s0 = wmma_bf16(qf[0], kf[0][0], z);
        s0     = wmma_bf16(qf[1], kf[0][1], s0);
        v8f s1 = wmma_bf16(qf[0], kf[1][0], z);
        s1     = wmma_bf16(qf[1], kf[1][1], s1);

        // additive position bias from LDS-staged tile
        const float* bw = &Bl[cur][wv][0];
#pragma unroll
        for (int r = 0; r < 8; ++r) {
            s0[r] += bw[(r + half * 8) * 32 + l15];
            s1[r] += bw[(r + half * 8) * 32 + 16 + l15];
        }

        // online softmax (rows live in 16-lane halves)
        float alpha[8];
#pragma unroll
        for (int r = 0; r < 8; ++r) {
            float mc = fmaxf(s0[r], s1[r]);
            mc = fmaxf(mc, __shfl_xor(mc, 1));
            mc = fmaxf(mc, __shfl_xor(mc, 2));
            mc = fmaxf(mc, __shfl_xor(mc, 4));
            mc = fmaxf(mc, __shfl_xor(mc, 8));
            float mnew = fmaxf(m[r], mc);
            alpha[r] = __expf(m[r] - mnew);
            float p0 = __expf(s0[r] - mnew);
            float p1 = __expf(s1[r] - mnew);
            s0[r] = p0; s1[r] = p1;
            float rs = p0 + p1;
            rs += __shfl_xor(rs, 1);
            rs += __shfl_xor(rs, 2);
            rs += __shfl_xor(rs, 4);
            rs += __shfl_xor(rs, 8);
            l[r] = l[r] * alpha[r] + rs;
            m[r] = mnew;
        }
#pragma unroll
        for (int j = 0; j < 4; ++j)
#pragma unroll
            for (int r = 0; r < 8; ++r) acc[j][r] *= alpha[r];

        // P (C-layout fp32) -> LDS bf16 16x32 -> reload in A-layout
#pragma unroll
        for (int r = 0; r < 8; ++r) {
            int row = r + half * 8;
            P[row * 32 + l15]      = f2bf(s0[r]);
            P[row * 32 + 16 + l15] = f2bf(s1[r]);
        }
        asm volatile("s_wait_dscnt 0" ::: "memory");   // same-wave LDS RAW fence
        v16bf pa;
        {
            FragU f;
            f.q[0] = *reinterpret_cast<const v4u*>(&P[l15 * 32 + klo]);
            f.q[1] = *reinterpret_cast<const v4u*>(&P[l15 * 32 + klo + 16]);
            pa = f.v;
        }

        // ctx += P[16x32] * V[32x64] (V from LDS, transposed layout)
#pragma unroll
        for (int j = 0; j < 4; ++j) {
            const unsigned short* vp = &Vl[cur][(j * 16 + l15) * VL_STRIDE];
            FragU f;
            f.q[0] = *reinterpret_cast<const v4u*>(&vp[klo]);
            f.q[1] = *reinterpret_cast<const v4u*>(&vp[klo + 16]);
            acc[j] = wmma_bf16(pa, f.v, acc[j]);
        }
    };

    stageAll(0, 0);
    wait_async0();
    __syncthreads();

    int cur = 0;
    for (int c = 0; c < NC - 1; ++c) {            // steady state: always stage
        stageAll(cur ^ 1, (c + 1) * 32);
        body(cur);
        wait_async0();
        __syncthreads();
        cur ^= 1;
    }
    body(cur);                                     // last chunk, no staging

    // normalize and store ctx -> [B,S,H*dk] bf16
#pragma unroll
    for (int j = 0; j < 4; ++j)
#pragma unroll
        for (int r = 0; r < 8; ++r) {
            float v = acc[j][r] / l[r];
            int row = qbase + r + half * 8;
            int col = h * DK + j * 16 + l15;
            Ctx[((size_t)(b * SS + row)) * DD + col] = f2bf(v);
        }
}

// ---------------------------------------------------------------------------
// Launch. Workspace (256-aligned, ~48.6 MB):
//   Xb 8M | WqT/WkT/WvT/WoT 2M ea | Qb 8M | Kb 8.25M | Vt 8.25M | Ctx 8M
// ---------------------------------------------------------------------------
extern "C" void kernel_launch(void* const* d_in, const int* in_sizes, int n_in,
                              void* d_out, int out_size, void* d_ws, size_t ws_size,
                              hipStream_t stream) {
    const float* hs   = (const float*)d_in[0];
    const float* pk   = (const float*)d_in[1];
    const float* pv   = (const float*)d_in[2];
    const float* bias = (const float*)d_in[3];
    const float* Wq   = (const float*)d_in[4];
    const float* Wk   = (const float*)d_in[5];
    const float* Wv   = (const float*)d_in[6];
    const float* Wo   = (const float*)d_in[7];
    float* out = (float*)d_out;

    char* ws = (char*)d_ws;
    size_t off = 0;
    auto alloc = [&](size_t bytes) -> void* {
        void* p = ws + off;
        off = (off + bytes + 255) & ~(size_t)255;
        return p;
    };
    unsigned short* Xb  = (unsigned short*)alloc((size_t)BS * DD * 2);
    unsigned short* WqT = (unsigned short*)alloc((size_t)DD * DD * 2);
    unsigned short* WkT = (unsigned short*)alloc((size_t)DD * DD * 2);
    unsigned short* WvT = (unsigned short*)alloc((size_t)DD * DD * 2);
    unsigned short* WoT = (unsigned short*)alloc((size_t)DD * DD * 2);
    unsigned short* Qb  = (unsigned short*)alloc((size_t)BB * HH * SS * DK * 2);
    unsigned short* Kb  = (unsigned short*)alloc((size_t)BB * HH * LL * DK * 2);
    unsigned short* Vt  = (unsigned short*)alloc((size_t)BB * HH * DK * LL * 2);
    unsigned short* Ctx = (unsigned short*)alloc((size_t)BS * DD * 2);

    const int nX = BS * DD;
    cast_bf16_kernel<<<(nX + 255) / 256, 256, 0, stream>>>(hs, Xb, nX);

    dim3 tg(DD / 32, DD / 32);   // (32, 32) tiles for weight transpose-cast
    castT_kernel<<<tg, 256, 0, stream>>>(Wq, WqT);
    castT_kernel<<<tg, 256, 0, stream>>>(Wk, WkT);
    castT_kernel<<<tg, 256, 0, stream>>>(Wv, WvT);
    castT_kernel<<<tg, 256, 0, stream>>>(Wo, WoT);

    const int nP = BB * HH * PP * DK;
    prompt_k_kernel<<<(nP + 255) / 256, 256, 0, stream>>>(pk, Kb);
    prompt_v_kernel<<<(nP + 255) / 256, 256, 0, stream>>>(pv, Vt);

    dim3 gg(DD / 128, BS / 128);   // (8, 32)
    gemm_bf16_kernel<<<gg, 256, 0, stream>>>(Xb, WqT, Qb, 0);
    gemm_bf16_kernel<<<gg, 256, 0, stream>>>(Xb, WkT, Kb, 1);
    gemm_bf16_kernel<<<gg, 256, 0, stream>>>(Xb, WvT, Vt, 2);

    attn_kernel<<<BB * HH * (SS / 64), 128, 0, stream>>>(Qb, Kb, Vt, bias, Ctx);

    gemm_bf16_kernel<<<gg, 256, 0, stream>>>(Ctx, WoT, out, 3);
}
